// PseudoTilinGNN_7971459301909
// MI455X (gfx1250) — compile-verified, matrix operand order checked by reading, not run on hardware
//
#include <hip/hip_runtime.h>
#include <hip/hip_bf16.h>
#include <math.h>

// ---------------------------------------------------------------------------
// CDNA5 (gfx1250) WMMA types & helpers
// ---------------------------------------------------------------------------
typedef __attribute__((ext_vector_type(16))) __bf16 v16bf;
typedef __attribute__((ext_vector_type(8)))  __bf16 v8bf;
typedef __attribute__((ext_vector_type(8)))  float  v8f;

__device__ __forceinline__ v8f wmma_bf16(v16bf a, v16bf b, v8f c) {
  // D = A(16x32 bf16) * B(32x16 bf16) + C(16x16 f32)
  return __builtin_amdgcn_wmma_f32_16x16x32_bf16(
      /*neg_a=*/false, a, /*neg_b=*/false, b,
      /*c_mod=*/(short)0, c, /*reuse_a=*/false, /*reuse_b=*/false);
}

// A fragment from global fp32 matrix [N x lda], rows row0..row0+15, k in [kBase, kBase+32)
// Per-lane k pattern: e<8 -> k = half*8+e ; e>=8 -> k = half*8+16+(e-8)
__device__ __forceinline__ v16bf load_A_g(const float* __restrict__ A, int row0,
                                          int lda, int kBase, int N) {
  int lane = threadIdx.x & 31;
  int m = lane & 15, half = lane >> 4;
  int r = row0 + m; if (r >= N) r = N - 1;           // clamp (keeps EXEC full)
  const float* p = A + (size_t)r * lda + kBase + half * 8;
  v16bf a;
#pragma unroll
  for (int t = 0; t < 8; ++t) { a[t] = (__bf16)p[t]; a[8 + t] = (__bf16)p[16 + t]; }
  return a;
}

// B fragment from global fp32 weight [K x ncols] row-major, col tile nBase
__device__ __forceinline__ v16bf load_B_g(const float* __restrict__ W, int ncols,
                                          int kBase, int nBase) {
  int lane = threadIdx.x & 31;
  int n = lane & 15, half = lane >> 4;
  const float* p = W + (size_t)(kBase + half * 8) * ncols + nBase + n;
  v16bf b;
#pragma unroll
  for (int t = 0; t < 8; ++t) {
    b[t]     = (__bf16)p[(size_t)t * ncols];
    b[8 + t] = (__bf16)p[(size_t)(16 + t) * ncols];
  }
  return b;
}

// A fragment from LDS bf16 staging [16 x ldS] (ldS*2 bytes multiple of 16)
__device__ __forceinline__ v16bf load_A_lds(const __bf16* S, int ldS, int kBase) {
  int lane = threadIdx.x & 31;
  int m = lane & 15, half = lane >> 4;
  const __bf16* p = S + m * ldS + kBase + half * 8;
  v8bf lo = *(const v8bf*)p;            // ds_load_b128
  v8bf hi = *(const v8bf*)(p + 16);     // ds_load_b128
  v16bf a;
#pragma unroll
  for (int t = 0; t < 8; ++t) { a[t] = lo[t]; a[8 + t] = hi[t]; }
  return a;
}

// Store a 16x16 f32 D tile into bf16 LDS staging with bias + LeakyReLU
__device__ __forceinline__ void store_stage(__bf16* S, int ldS, int nBase, v8f c,
                                            const float* __restrict__ bias) {
  int lane = threadIdx.x & 31;
  int n = lane & 15, half = lane >> 4;
  float bb = bias[nBase + n];
#pragma unroll
  for (int j = 0; j < 8; ++j) {
    int m = j + 8 * half;
    float v = c[j] + bb;
    v = v > 0.f ? v : 0.01f * v;
    S[m * ldS + nBase + n] = (__bf16)v;
  }
}

// ---------------------------------------------------------------------------
// Elementwise / reduction kernels
// ---------------------------------------------------------------------------
__global__ void k_zero(float* __restrict__ p, long n) {
  long i = (long)blockIdx.x * blockDim.x + threadIdx.x;
  if (i < n) p[i] = 0.f;
}

// y[n][j] = x[n] * Wi0[j] + bi0[j]   (first init layer, K=1 -> no WMMA needed)
__global__ void k_init0(const float* __restrict__ x, const float* __restrict__ W,
                        const float* __restrict__ b, float* __restrict__ y, int N) {
  long i = (long)blockIdx.x * blockDim.x + threadIdx.x;
  if (i >= (long)N * 32) return;
  int c = (int)(i & 31);
  long r = i >> 5;
  y[i] = x[r] * W[c] + b[c];
}

// Per-column sum / sum-of-squares over N rows of a [N x 32] matrix (fp32)
__global__ void k_colstats(const float* __restrict__ Y, float* __restrict__ stats, int N) {
  __shared__ float s_sum[32], s_sq[32];
  if (threadIdx.x < 32) { s_sum[threadIdx.x] = 0.f; s_sq[threadIdx.x] = 0.f; }
  __syncthreads();
  int col = threadIdx.x & 31;
  int rowBase = blockIdx.x * 1024 + (threadIdx.x >> 5);
  float sum = 0.f, sq = 0.f;
#pragma unroll 4
  for (int r = 0; r < 128; ++r) {
    int row = rowBase + r * 8;
    if (row < N) { float v = Y[(size_t)row * 32 + col]; sum += v; sq += v * v; }
  }
  atomicAdd(&s_sum[col], sum);
  atomicAdd(&s_sq[col], sq);
  __syncthreads();
  if (threadIdx.x < 32) {
    atomicAdd(&stats[threadIdx.x],      s_sum[threadIdx.x]);
    atomicAdd(&stats[32 + threadIdx.x], s_sq[threadIdx.x]);
  }
}

// scale = g * rsqrt(var+eps), shift = be - mean*scale   (32 threads)
__global__ void k_finalize(const float* __restrict__ stats, const float* __restrict__ g,
                           const float* __restrict__ be, float* __restrict__ scale,
                           float* __restrict__ shift, int N) {
  int c = threadIdx.x;
  float invN = 1.f / (float)N;
  float mean = stats[c] * invN;
  float var  = stats[32 + c] * invN - mean * mean;
  float sc   = g[c] * rsqrtf(var + 1e-5f);
  scale[c] = sc;
  shift[c] = be[c] - mean * sc;
}

// out = lrelu(y*scale+shift) [+ resid]
__global__ void k_bn_act(const float* __restrict__ Y, const float* __restrict__ scale,
                         const float* __restrict__ shift, const float* __restrict__ resid,
                         float* __restrict__ out, int N) {
  long i = (long)blockIdx.x * blockDim.x + threadIdx.x;
  if (i >= (long)N * 32) return;
  int c = (int)(i & 31);
  float v = Y[i] * scale[c] + shift[c];
  v = v > 0.f ? v : 0.01f * v;
  if (resid) v += resid[i];
  out[i] = v;
}

// agg[dst] += msg[src] : one wave per edge, one lane per feature (coalesced 128B)
__global__ void k_edge(const float* __restrict__ msg, const int* __restrict__ src,
                       const int* __restrict__ dst, float* __restrict__ agg, int E) {
  long tid = (long)blockIdx.x * blockDim.x + threadIdx.x;
  int e = (int)(tid >> 5);
  if (e >= E) return;
  int f = (int)(tid & 31);
  int s = src[e], d = dst[e];
  atomicAdd(&agg[(size_t)d * 32 + f], msg[(size_t)s * 32 + f]);
}

// ---------------------------------------------------------------------------
// WMMA GEMM: Y[N x 32] = A1[N x 32] @ B1[32 x 32] (+ A2 @ B2) + bias
// One wave per 16-row tile; two 16x16 output tiles (K=32 -> one WMMA each).
// ---------------------------------------------------------------------------
__global__ __launch_bounds__(256)
void k_gemm32(const float* __restrict__ A1, const float* __restrict__ B1,
              const float* __restrict__ A2, const float* __restrict__ B2,
              const float* __restrict__ bias, float* __restrict__ Y, int N) {
  int wave = (int)(((long)blockIdx.x * blockDim.x + threadIdx.x) >> 5);
  int row0 = wave * 16;
  if (row0 >= N) return;                 // wave-uniform: EXEC stays full for WMMA
  int lane = threadIdx.x & 31;
  v8f c0 = {}, c1 = {};
  {
    v16bf a  = load_A_g(A1, row0, 32, 0, N);
    v16bf b0 = load_B_g(B1, 32, 0, 0);
    v16bf b1 = load_B_g(B1, 32, 0, 16);
    c0 = wmma_bf16(a, b0, c0);
    c1 = wmma_bf16(a, b1, c1);
  }
  if (A2) {                              // scalar-pointer branch: wave-uniform
    v16bf a  = load_A_g(A2, row0, 32, 0, N);
    v16bf b0 = load_B_g(B2, 32, 0, 0);
    v16bf b1 = load_B_g(B2, 32, 0, 16);
    c0 = wmma_bf16(a, b0, c0);
    c1 = wmma_bf16(a, b1, c1);
  }
  int n = lane & 15, half = lane >> 4;
  float bb0 = bias[n], bb1 = bias[16 + n];
#pragma unroll
  for (int j = 0; j < 8; ++j) {
    int r = row0 + j + 8 * half;
    if (r < N) {
      Y[(size_t)r * 32 + n]      = c0[j] + bb0;
      Y[(size_t)r * 32 + 16 + n] = c1[j] + bb1;
    }
  }
}

// ---------------------------------------------------------------------------
// Fused final MLP: 32 -> 256 -> 128 -> 64 -> 32 -> sigmoid(1)
// One wave per 16-row tile; activations staged per-wave in LDS as bf16
// (padded row stride => conflict-free ds_load_b128). Weights broadcast from L2.
// ---------------------------------------------------------------------------
#define FM_WAVES 4
__global__ __launch_bounds__(128)
void k_final_mlp(const float* __restrict__ H,
                 const float* __restrict__ Wf0, const float* __restrict__ bf0,
                 const float* __restrict__ Wf1, const float* __restrict__ bf1,
                 const float* __restrict__ Wf2, const float* __restrict__ bf2,
                 const float* __restrict__ Wf3, const float* __restrict__ bf3,
                 const float* __restrict__ Wo,  const float* __restrict__ bo,
                 float* __restrict__ out, int N) {
  __shared__ __align__(16) __bf16 stg1[FM_WAVES][16][264];  // a0(256) / a2(64)
  __shared__ __align__(16) __bf16 stg2[FM_WAVES][16][136];  // a1(128) / a3(32)
  int w = threadIdx.x >> 5;
  int lane = threadIdx.x & 31;
  int row0 = (blockIdx.x * FM_WAVES + w) * 16;
  __bf16* S1 = &stg1[w][0][0];
  __bf16* S2 = &stg2[w][0][0];

  // ---- stage 0: a0 = lrelu(h @ Wf0 + bf0), 16 col tiles, K=32
  {
    v16bf a = load_A_g(H, row0, 32, 0, N);
#pragma unroll
    for (int t = 0; t < 16; ++t) {
      v16bf b = load_B_g(Wf0, 256, 0, t * 16);
      v8f c = {};
      c = wmma_bf16(a, b, c);
      store_stage(S1, 264, t * 16, c, bf0);
    }
  }
  __syncthreads();
  // ---- stage 1: a1 = lrelu(a0 @ Wf1 + bf1), 8 col tiles x 8 K-steps
#pragma unroll
  for (int t = 0; t < 8; ++t) {
    v8f c = {};
#pragma unroll
    for (int s = 0; s < 8; ++s) {
      v16bf a = load_A_lds(S1, 264, s * 32);
      v16bf b = load_B_g(Wf1, 128, s * 32, t * 16);
      c = wmma_bf16(a, b, c);
    }
    store_stage(S2, 136, t * 16, c, bf1);
  }
  __syncthreads();
  // ---- stage 2: a2 = lrelu(a1 @ Wf2 + bf2), 4 col tiles x 4 K-steps
#pragma unroll
  for (int t = 0; t < 4; ++t) {
    v8f c = {};
#pragma unroll
    for (int s = 0; s < 4; ++s) {
      v16bf a = load_A_lds(S2, 136, s * 32);
      v16bf b = load_B_g(Wf2, 64, s * 32, t * 16);
      c = wmma_bf16(a, b, c);
    }
    store_stage(S1, 264, t * 16, c, bf2);
  }
  __syncthreads();
  // ---- stage 3: a3 = lrelu(a2 @ Wf3 + bf3), 2 col tiles x 2 K-steps
#pragma unroll
  for (int t = 0; t < 2; ++t) {
    v8f c = {};
#pragma unroll
    for (int s = 0; s < 2; ++s) {
      v16bf a = load_A_lds(S1, 264, s * 32);
      v16bf b = load_B_g(Wf3, 32, s * 32, t * 16);
      c = wmma_bf16(a, b, c);
    }
    store_stage(S2, 136, t * 16, c, bf3);
  }
  __syncthreads();
  // ---- head: out = sigmoid(a3 . Wo + bo); lane pairs (L, L+16) split K=32
  {
    int m = lane & 15, half = lane >> 4;
    float sum = 0.f;
    const __bf16* p = S2 + m * 136 + half * 16;
#pragma unroll
    for (int k = 0; k < 16; ++k) sum += (float)p[k] * Wo[half * 16 + k];
    sum += __shfl_xor(sum, 16, 32);
    int r = row0 + m;
    if (half == 0 && r < N) {
      float z = sum + bo[0];
      out[r] = 1.f / (1.f + expf(-z));
    }
  }
}

// ---------------------------------------------------------------------------
// Host orchestration
// ---------------------------------------------------------------------------
extern "C" void kernel_launch(void* const* d_in, const int* in_sizes, int n_in,
                              void* d_out, int out_size, void* d_ws, size_t ws_size,
                              hipStream_t stream) {
  (void)n_in; (void)out_size; (void)ws_size;
  const float* x    = (const float*)d_in[0];
  const int*   cei  = (const int*)  d_in[1];
  const float* Wi0  = (const float*)d_in[2];
  const float* bi0  = (const float*)d_in[3];
  const float* gi0  = (const float*)d_in[4];
  const float* bei0 = (const float*)d_in[5];
  const float* Wi1  = (const float*)d_in[6];
  const float* bi1  = (const float*)d_in[7];
  const float* gi1  = (const float*)d_in[8];
  const float* bei1 = (const float*)d_in[9];
  const float* Wm   = (const float*)d_in[10];
  const float* bm   = (const float*)d_in[11];
  const float* gm   = (const float*)d_in[12];
  const float* bem  = (const float*)d_in[13];
  const float* Wu   = (const float*)d_in[14];
  const float* bu   = (const float*)d_in[15];
  const float* gu   = (const float*)d_in[16];
  const float* beu  = (const float*)d_in[17];
  const float* Wf0  = (const float*)d_in[18];
  const float* bf0  = (const float*)d_in[19];
  const float* Wf1  = (const float*)d_in[20];
  const float* bf1  = (const float*)d_in[21];
  const float* Wf2  = (const float*)d_in[22];
  const float* bf2  = (const float*)d_in[23];
  const float* Wf3  = (const float*)d_in[24];
  const float* bf3  = (const float*)d_in[25];
  const float* Wo   = (const float*)d_in[26];
  const float* bo   = (const float*)d_in[27];

  const int N = in_sizes[0];
  const int E = in_sizes[1] / 2;
  const int D = in_sizes[10] / (32 * 32);
  const int* src = cei;
  const int* dst = cei + E;

  const size_t NW = (size_t)N * 32;
  float* ws = (float*)d_ws;
  float* ring[4] = { ws, ws + NW, ws + 2 * NW, ws + 3 * NW };  // feats ring (skip=2)
  float* yb    = ws + 4 * NW;   // pre-BN GEMM output
  float* msg   = ws + 5 * NW;   // message features
  float* agg   = ws + 6 * NW;   // scatter-sum accumulator
  float* stats = ws + 7 * NW;   // [64] sums  + [32] scale + [32] shift
  float* scale = stats + 64;
  float* shift = stats + 96;

  const int ewB   = (int)(((long)N * 32 + 255) / 256);           // elementwise grid
  const int csB   = (N + 1023) / 1024;                           // colstats grid
  const int gemmB = (int)(((long)((N + 15) / 16) * 32 + 255) / 256);
  const int edgeB = (int)(((long)E * 32 + 255) / 256);

  // ---- init MLP layer 0: Linear(1->32) -> BN -> LeakyReLU
  k_zero<<<1, 128, 0, stream>>>(stats, 128);
  k_init0<<<ewB, 256, 0, stream>>>(x, Wi0, bi0, yb, N);
  k_colstats<<<csB, 256, 0, stream>>>(yb, stats, N);
  k_finalize<<<1, 32, 0, stream>>>(stats, gi0, bei0, scale, shift, N);
  k_bn_act<<<ewB, 256, 0, stream>>>(yb, scale, shift, nullptr, ring[0], N);

  // ---- init MLP layer 1: Linear(32->32) -> BN -> LeakyReLU
  k_zero<<<1, 128, 0, stream>>>(stats, 128);
  k_gemm32<<<gemmB, 256, 0, stream>>>(ring[0], Wi1, nullptr, nullptr, bi1, yb, N);
  k_colstats<<<csB, 256, 0, stream>>>(yb, stats, N);
  k_finalize<<<1, 32, 0, stream>>>(stats, gi1, bei1, scale, shift, N);
  k_bn_act<<<ewB, 256, 0, stream>>>(yb, scale, shift, nullptr, ring[0], N);

  // ---- D CollConv layers with residual skip of 2
  for (int i = 0; i < D; ++i) {
    const float* Wm_i  = Wm + (size_t)i * 1024;
    const float* bm_i  = bm + (size_t)i * 32;
    const float* gm_i  = gm + (size_t)i * 32;
    const float* bem_i = bem + (size_t)i * 32;
    const float* Wu_i  = Wu + (size_t)i * 2048;   // [64 x 32]: rows 0-31 = h, 32-63 = agg
    const float* bu_i  = bu + (size_t)i * 32;
    const float* gu_i  = gu + (size_t)i * 32;
    const float* beu_i = beu + (size_t)i * 32;

    float* h     = ring[i & 3];
    float* outb  = ring[(i + 1) & 3];
    const float* resid = (i >= 2) ? ring[(i - 2) & 3] : nullptr;

    // msg = lrelu(bn(h @ Wm + bm))
    k_zero<<<1, 128, 0, stream>>>(stats, 128);
    k_gemm32<<<gemmB, 256, 0, stream>>>(h, Wm_i, nullptr, nullptr, bm_i, yb, N);
    k_colstats<<<csB, 256, 0, stream>>>(yb, stats, N);
    k_finalize<<<1, 32, 0, stream>>>(stats, gm_i, bem_i, scale, shift, N);
    k_bn_act<<<ewB, 256, 0, stream>>>(yb, scale, shift, nullptr, msg, N);

    // agg = segment_sum(msg[src], dst)
    k_zero<<<ewB, 256, 0, stream>>>(agg, (long)NW);
    k_edge<<<edgeB, 256, 0, stream>>>(msg, src, dst, agg, E);

    // h' = lrelu(bn([h, agg] @ Wu + bu)) [+ feats[i-2]]
    k_zero<<<1, 128, 0, stream>>>(stats, 128);
    k_gemm32<<<gemmB, 256, 0, stream>>>(h, Wu_i, agg, Wu_i + 1024, bu_i, yb, N);
    k_colstats<<<csB, 256, 0, stream>>>(yb, stats, N);
    k_finalize<<<1, 32, 0, stream>>>(stats, gu_i, beu_i, scale, shift, N);
    k_bn_act<<<ewB, 256, 0, stream>>>(yb, scale, shift, resid, outb, N);
  }

  // ---- fused final MLP + sigmoid head
  float* hfin = ring[D & 3];
  k_final_mlp<<<(N + 16 * FM_WAVES - 1) / (16 * FM_WAVES), 32 * FM_WAVES, 0, stream>>>(
      hfin, Wf0, bf0, Wf1, bf1, Wf2, bf2, Wf3, bf3, Wo, bo, (float*)d_out, N);
}
